// CausalSelfAttention_12189117186204
// MI455X (gfx1250) — matrix operand order repaired and checked by reading, not compiled, not run on hardware
//
#include <hip/hip_runtime.h>
#include <hip/hip_bf16.h>
#include <stdint.h>

// ---------------------------------------------------------------------------
// CDNA5 (gfx1250) causal self-attention: convert -> QKV gemm -> flash attn
// -> proj. All matmuls on v_wmma_f32_16x16x32_bf16 (wave32), fp32 accum.
// All fp32->bf16 conversion hoisted out of GEMM hot loops; fragments load as
// b128 from pre-converted bf16 operands.
// ---------------------------------------------------------------------------

typedef __bf16 v16bf __attribute__((ext_vector_type(16)));
typedef float  v8f   __attribute__((ext_vector_type(8)));

union Frag { v16bf v; uint32_t u[8]; };

__device__ __forceinline__ uint32_t pk_bf16(float x, float y) {
  union { __bf16 h[2]; uint32_t u; } r;
  r.h[0] = (__bf16)x; r.h[1] = (__bf16)y;
  return r.u;
}
__device__ __forceinline__ uint16_t bf_bits(float x) {
  union { __bf16 h; uint16_t u; } r;
  r.h = (__bf16)x;
  return r.u;
}
__device__ __forceinline__ void load_b128(uint32_t* dst, const uint16_t* src) {
  *(uint4*)dst = *(const uint4*)src;
}

__device__ __forceinline__ v8f wmma_bf16(const Frag& a, const Frag& b, v8f c) {
  return __builtin_amdgcn_wmma_f32_16x16x32_bf16(false, a.v, false, b.v,
                                                 (short)0, c, false, false);
}

#define SEQ    2048
#define DMODEL 1024
#define NHEAD  16
#define HDIM   64

// ---------------------------------------------------------------------------
// Kernel 0: elementwise fp32 -> bf16 (8 elems/thread, b128 in / b128 out)
// ---------------------------------------------------------------------------
__global__ __launch_bounds__(256)
void cvt_bf16(const float* __restrict__ in, uint16_t* __restrict__ out) {
  size_t i = ((size_t)blockIdx.x * 256 + threadIdx.x) * 8;
  float4 f0 = *(const float4*)(in + i);
  float4 f1 = *(const float4*)(in + i + 4);
  uint4 o;
  o.x = pk_bf16(f0.x, f0.y);
  o.y = pk_bf16(f0.z, f0.w);
  o.z = pk_bf16(f1.x, f1.y);
  o.w = pk_bf16(f1.z, f1.w);
  *(uint4*)(out + i) = o;
}

// ---------------------------------------------------------------------------
// Kernel 1: qkv = x @ w_qkv^T   (M=8192, N=3072, K=1024), bf16 operands.
// Per-wave tile 32x64 (2x4 WMMA frags), workgroup 8 waves = 64x256 tile.
// Scatter-stores bf16 Q (prescaled 1/8), K [BH,S,64], V^T [BH,64,S].
// ---------------------------------------------------------------------------
__global__ __launch_bounds__(256)
void qkv_gemm(const uint16_t* __restrict__ xb, const uint16_t* __restrict__ wb,
              uint16_t* __restrict__ qws, uint16_t* __restrict__ kws,
              uint16_t* __restrict__ vtws) {
  const int lane = threadIdx.x & 31;
  const int wv   = threadIdx.x >> 5;
  const int wm   = wv >> 2, wn = wv & 3;
  const int half = lane >> 4, l16 = lane & 15;
  const int m_base = blockIdx.y * 64  + wm * 32;
  const int n_base = blockIdx.x * 256 + wn * 64;

  const uint16_t* arow0 = xb + (size_t)(m_base + l16) * DMODEL;      // i=0
  const uint16_t* arow1 = arow0 + (size_t)16 * DMODEL;               // i=1
  const uint16_t* brow[4];
#pragma unroll
  for (int j = 0; j < 4; j++)
    brow[j] = wb + (size_t)(n_base + j * 16 + l16) * DMODEL;

  v8f acc[2][4];
  for (int i = 0; i < 2; i++)
    for (int j = 0; j < 4; j++)
      for (int e = 0; e < 8; e++) acc[i][j][e] = 0.f;

#pragma unroll 2
  for (int k0 = 0; k0 < DMODEL; k0 += 32) {
    Frag a[2], b[4];
    // A: lane's v0..3 = K half*8+0..7, v4..7 = K 16+half*8+0..7 (contiguous)
    load_b128(&a[0].u[0], arow0 + k0 + half * 8);
    load_b128(&a[0].u[4], arow0 + k0 + 16 + half * 8);
    load_b128(&a[1].u[0], arow1 + k0 + half * 8);
    load_b128(&a[1].u[4], arow1 + k0 + 16 + half * 8);
    // B: lane's v0..3 = K half*8+0..7, v4..7 = K 16+half*8+0..7
#pragma unroll
    for (int j = 0; j < 4; j++) {
      load_b128(&b[j].u[0], brow[j] + k0 + half * 8);
      load_b128(&b[j].u[4], brow[j] + k0 + 16 + half * 8);
    }
#pragma unroll
    for (int i = 0; i < 2; i++)
#pragma unroll
      for (int j = 0; j < 4; j++)
        acc[i][j] = wmma_bf16(a[i], b[j], acc[i][j]);
  }

#pragma unroll
  for (int i = 0; i < 2; i++)
#pragma unroll
    for (int j = 0; j < 4; j++)
#pragma unroll
      for (int r = 0; r < 8; r++) {
        int row = m_base + i * 16 + r + half * 8;   // 0..8191
        int col = n_base + j * 16 + l16;            // 0..3071
        float val = acc[i][j][r];
        int bb = row >> 11, s = row & (SEQ - 1);
        int which = col >> 10, d = col & (DMODEL - 1);
        int h = d >> 6, hd = d & (HDIM - 1);
        int bh = bb * NHEAD + h;
        if (which == 0)
          qws[((size_t)bh * SEQ + s) * HDIM + hd] = bf_bits(val * 0.125f);
        else if (which == 1)
          kws[((size_t)bh * SEQ + s) * HDIM + hd] = bf_bits(val);
        else
          vtws[((size_t)bh * HDIM + hd) * SEQ + s] = bf_bits(val);
      }
}

// ---------------------------------------------------------------------------
// Kernel 2: flash attention. 8 waves/block, each wave owns 16 queries.
// Per 32-key tile: 4 WMMA (QK^T, K=64), online softmax, P via LDS transpose,
// 4 WMMA (P*V) into fp32 O accumulators. Causal: fully-masked tiles skipped.
// ---------------------------------------------------------------------------
__global__ __launch_bounds__(256)
void attn_kernel(const uint16_t* __restrict__ qws, const uint16_t* __restrict__ kws,
                 const uint16_t* __restrict__ vtws, uint16_t* __restrict__ attn) {
  __shared__ uint16_t p_lds[8][16][32];   // 1KB per wave
  const int lane = threadIdx.x & 31;
  const int wv   = threadIdx.x >> 5;
  const int half = lane >> 4, l16 = lane & 15;
  const int bh = blockIdx.x >> 4;           // S/128 = 16 q-blocks
  const int qb = blockIdx.x & 15;
  const int q0 = qb * 128 + wv * 16;

  Frag aq[2];                               // Q rows, head-dim 0..31 / 32..63
  {
    const uint16_t* qrow = qws + ((size_t)bh * SEQ + q0 + l16) * HDIM;
#pragma unroll
    for (int p = 0; p < 2; p++) {
      load_b128(&aq[p].u[0], qrow + p * 32 + half * 8);
      load_b128(&aq[p].u[4], qrow + p * 32 + 16 + half * 8);
    }
  }

  float mrow[8], lrow[8];
  v8f o[4];
#pragma unroll
  for (int r = 0; r < 8; r++) { mrow[r] = -INFINITY; lrow[r] = 0.f; }
  for (int t = 0; t < 4; t++)
    for (int e = 0; e < 8; e++) o[t][e] = 0.f;

  const int ntiles = (q0 + 16 + 31) >> 5;   // keys 0 .. q0+15
  for (int tix = 0; tix < ntiles; tix++) {
    const int kt = tix * 32;

    // --- S = Q * K^T (two 16-key column tiles) ---
    Frag bk[2][2];
#pragma unroll
    for (int kh = 0; kh < 2; kh++) {
      const uint16_t* krow = kws + ((size_t)bh * SEQ + kt + kh * 16 + l16) * HDIM;
#pragma unroll
      for (int p = 0; p < 2; p++) {
        load_b128(&bk[kh][p].u[0], krow + p * 32 + half * 8);
        load_b128(&bk[kh][p].u[4], krow + p * 32 + 16 + half * 8);
      }
    }
    v8f s[2];
#pragma unroll
    for (int kh = 0; kh < 2; kh++) {
      v8f z; for (int e = 0; e < 8; e++) z[e] = 0.f;
      z = wmma_bf16(aq[0], bk[kh][0], z);
      z = wmma_bf16(aq[1], bk[kh][1], z);
      s[kh] = z;
    }

    // --- causal mask (boundary tile only; wave-uniform branch) ---
    if (kt + 31 > q0) {
#pragma unroll
      for (int r = 0; r < 8; r++) {
        int qrow_i = q0 + r + half * 8;
        int key0 = kt + l16;
        if (key0 > qrow_i)      s[0][r] = -INFINITY;
        if (key0 + 16 > qrow_i) s[1][r] = -INFINITY;
      }
    }

    // --- online softmax (rows striped across 16-lane halves) ---
    float p0[8], p1[8], alpha[8];
#pragma unroll
    for (int r = 0; r < 8; r++) {
      float mx = fmaxf(s[0][r], s[1][r]);
      for (int off = 8; off; off >>= 1) mx = fmaxf(mx, __shfl_xor(mx, off, 32));
      float mn = fmaxf(mrow[r], mx);
      alpha[r] = __expf(mrow[r] - mn);
      mrow[r] = mn;
      p0[r] = __expf(s[0][r] - mn);
      p1[r] = __expf(s[1][r] - mn);
      float rs = p0[r] + p1[r];
      for (int off = 8; off; off >>= 1) rs += __shfl_xor(rs, off, 32);
      lrow[r] = lrow[r] * alpha[r] + rs;
    }
#pragma unroll
    for (int t = 0; t < 4; t++)
#pragma unroll
      for (int r = 0; r < 8; r++) o[t][r] *= alpha[r];

    // --- P: C-layout -> A-layout via per-wave LDS tile ---
#pragma unroll
    for (int r = 0; r < 8; r++) {
      int m = r + half * 8;
      p_lds[wv][m][l16]      = bf_bits(p0[r]);
      p_lds[wv][m][l16 + 16] = bf_bits(p1[r]);
    }
    asm volatile("s_wait_dscnt 0x0" ::: "memory");  // per-wave LDS RAW fence
    Frag pf;
    load_b128(&pf.u[0], &p_lds[wv][l16][half * 8]);
    load_b128(&pf.u[4], &p_lds[wv][l16][16 + half * 8]);

    // --- O += P * V (V^T layout -> contiguous B-frag loads) ---
#pragma unroll
    for (int t = 0; t < 4; t++) {
      Frag bv;
      const uint16_t* vrow = vtws + ((size_t)bh * HDIM + t * 16 + l16) * SEQ;
      load_b128(&bv.u[0], vrow + kt + half * 8);
      load_b128(&bv.u[4], vrow + kt + 16 + half * 8);
      o[t] = wmma_bf16(pf, bv, o[t]);
    }
  }

  // --- finalize: O /= l, store bf16 [B,S,D] ---
  const int b = bh >> 4, h = bh & 15;
#pragma unroll
  for (int r = 0; r < 8; r++) {
    float inv = 1.0f / lrow[r];
    int q = q0 + r + half * 8;
    uint16_t* orow = attn + ((size_t)b * SEQ + q) * DMODEL + h * HDIM;
#pragma unroll
    for (int t = 0; t < 4; t++)
      orow[t * 16 + l16] = bf_bits(o[t][r] * inv);
  }
}

// ---------------------------------------------------------------------------
// Kernel 3: out = attn @ w_proj^T  (M=8192, N=1024, K=1024), fp32 output.
// ---------------------------------------------------------------------------
__global__ __launch_bounds__(256)
void proj_gemm(const uint16_t* __restrict__ a_bf, const uint16_t* __restrict__ wb,
               float* __restrict__ out) {
  const int lane = threadIdx.x & 31;
  const int wv   = threadIdx.x >> 5;
  const int wm   = wv >> 2, wn = wv & 3;
  const int half = lane >> 4, l16 = lane & 15;
  const int m_base = blockIdx.y * 64  + wm * 32;
  const int n_base = blockIdx.x * 256 + wn * 64;

  const uint16_t* arow0 = a_bf + (size_t)(m_base + l16) * DMODEL;
  const uint16_t* arow1 = arow0 + (size_t)16 * DMODEL;
  const uint16_t* brow[4];
#pragma unroll
  for (int j = 0; j < 4; j++)
    brow[j] = wb + (size_t)(n_base + j * 16 + l16) * DMODEL;

  v8f acc[2][4];
  for (int i = 0; i < 2; i++)
    for (int j = 0; j < 4; j++)
      for (int e = 0; e < 8; e++) acc[i][j][e] = 0.f;

#pragma unroll 2
  for (int k0 = 0; k0 < DMODEL; k0 += 32) {
    Frag a[2], b[4];
    load_b128(&a[0].u[0], arow0 + k0 + half * 8);
    load_b128(&a[0].u[4], arow0 + k0 + 16 + half * 8);
    load_b128(&a[1].u[0], arow1 + k0 + half * 8);
    load_b128(&a[1].u[4], arow1 + k0 + 16 + half * 8);
#pragma unroll
    for (int j = 0; j < 4; j++) {
      load_b128(&b[j].u[0], brow[j] + k0 + half * 8);
      load_b128(&b[j].u[4], brow[j] + k0 + 16 + half * 8);
    }
#pragma unroll
    for (int i = 0; i < 2; i++)
#pragma unroll
      for (int j = 0; j < 4; j++)
        acc[i][j] = wmma_bf16(a[i], b[j], acc[i][j]);
  }

#pragma unroll
  for (int i = 0; i < 2; i++)
#pragma unroll
    for (int j = 0; j < 4; j++)
#pragma unroll
      for (int r = 0; r < 8; r++) {
        int row = m_base + i * 16 + r + half * 8;
        int col = n_base + j * 16 + l16;
        out[(size_t)row * DMODEL + col] = acc[i][j][r];
      }
}

// ---------------------------------------------------------------------------
extern "C" void kernel_launch(void* const* d_in, const int* in_sizes, int n_in,
                              void* d_out, int out_size, void* d_ws, size_t ws_size,
                              hipStream_t stream) {
  const float* x      = (const float*)d_in[0];   // [4,2048,1024]
  const float* w_qkv  = (const float*)d_in[1];   // [3072,1024]
  const float* w_proj = (const float*)d_in[2];   // [1024,1024]
  float* out = (float*)d_out;                    // [4,2048,1024] fp32

  const size_t XE  = (size_t)4 * SEQ * DMODEL;       // 8388608
  const size_t WQE = (size_t)3 * DMODEL * DMODEL;    // 3145728
  const size_t WPE = (size_t)DMODEL * DMODEL;        // 1048576
  const size_t QKV = (size_t)4 * NHEAD * SEQ * HDIM; // 8388608

  uint16_t* ws   = (uint16_t*)d_ws;
  uint16_t* xb   = ws;                       // bf16 x [B*S, D]
  uint16_t* wqb  = xb + XE;                  // bf16 w_qkv [3D, D]
  uint16_t* wpb  = wqb + WQE;                // bf16 w_proj [D, D]
  uint16_t* qws  = wpb + WPE;                // bf16 [BH,S,64], pre-scaled 1/8
  uint16_t* kws  = qws + QKV;                // bf16 [BH,S,64]
  uint16_t* vtws = kws + QKV;                // bf16 [BH,64,S] (V transposed)
  uint16_t* attn = vtws + QKV;               // bf16 [B,S,D]

  cvt_bf16<<<dim3(XE  / (256 * 8)), 256, 0, stream>>>(x, xb);       // 4096
  cvt_bf16<<<dim3(WQE / (256 * 8)), 256, 0, stream>>>(w_qkv, wqb);  // 1536
  cvt_bf16<<<dim3(WPE / (256 * 8)), 256, 0, stream>>>(w_proj, wpb); // 512

  qkv_gemm<<<dim3(12, 128), 256, 0, stream>>>(xb, wqb, qws, kws, vtws);
  attn_kernel<<<dim3(64 * 16), 256, 0, stream>>>(qws, kws, vtws, attn);
  proj_gemm<<<dim3(4, 128), 256, 0, stream>>>(attn, wpb, out);
}
